// TransformerTTS_6373731467704
// MI455X (gfx1250) — matrix operand order repaired and checked by reading, compiled, not run on hardware
//
#include <hip/hip_runtime.h>
#include <hip/hip_bf16.h>

// ---------------------------------------------------------------------------
// TransformerTTS forward for gfx1250 (MI455X).
// All GEMMs/attention run on v_wmma_f32_16x16x32_bf16 with bf16 operands that
// are staged in workspace once (weights) or produced directly in bf16 by the
// upstream kernel (activations). f32 masters kept only for residual/LayerNorm.
// GEMM: 64x32 strip per wave, register double-buffered K loop.
// All hot-loop addressing: uniform SGPR base + 32-bit per-lane offsets (SADDR).
// ---------------------------------------------------------------------------

typedef __bf16 bf16_t;
typedef __attribute__((ext_vector_type(16))) __bf16 bf16x16;
typedef __attribute__((ext_vector_type(8)))  __bf16 bf16x8;
typedef __attribute__((ext_vector_type(4)))  __bf16 bf16x4;
typedef __attribute__((ext_vector_type(8)))  float  f32x8;
typedef __attribute__((ext_vector_type(4)))  float  f32x4;

#define TTS_B   4
#define TTS_S   512
#define TTS_E   512
#define TTS_H   8
#define TTS_L   4
#define TTS_HID 2048
#define TTS_OUT 80
#define TTS_M   1024
#define TTS_T   2048
#define TTS_HD  64

__device__ __forceinline__ f32x8 zero8() {
  f32x8 z = {0.f, 0.f, 0.f, 0.f, 0.f, 0.f, 0.f, 0.f};
  return z;
}

__device__ __forceinline__ f32x8 wmma_bf16(bf16x16 a, bf16x16 b, f32x8 c) {
  return __builtin_amdgcn_wmma_f32_16x16x32_bf16(
      false, a, false, b, (short)0, c, false, false);
}

__device__ __forceinline__ bf16x16 cat8(bf16x8 lo, bf16x8 hi) {
  return __builtin_shufflevector(lo, hi, 0, 1, 2, 3, 4, 5, 6, 7,
                                 8, 9, 10, 11, 12, 13, 14, 15);
}

// A-operand fragment (16x32 MxK). off (32-bit) = row*lda + half*16 + k.
// e0..15 <- base[off..off+15]  (contiguous k, matches B fragment).
__device__ __forceinline__ bf16x16 load_a_frag(const bf16_t* base, int off) {
  bf16x8 lo = *reinterpret_cast<const bf16x8*>(base + off);
  bf16x8 hi = *reinterpret_cast<const bf16x8*>(base + off + 8);
  return cat8(lo, hi);
}

// B-operand fragment (32x16 KxN), one column per lane.
// off (32-bit) = col*K + half*16 + k; 16 contiguous bf16.
__device__ __forceinline__ bf16x16 load_b_frag(const bf16_t* base, int off) {
  bf16x8 lo = *reinterpret_cast<const bf16x8*>(base + off);
  bf16x8 hi = *reinterpret_cast<const bf16x8*>(base + off + 8);
  return cat8(lo, hi);
}

// ---------------------------------------------------------------------------
// GEMM: C[M,N] = act(A[M,K](bf16) @ W[N,K](bf16)^T + bias(f32)).
// 256 threads = 8 waves; each wave owns a 64 x (NREP*16) strip:
// NREP weight fragments x 4 row fragments -> 4*NREP WMMA per k step.
// K-loop is register double-buffered so WMMA overlaps the next loads.
// N = gridDim.x*16*NREP, M covered by gridDim.y*512. Cf/Cb either may be null.
// ---------------------------------------------------------------------------
template <int NREP>
__global__ __launch_bounds__(256)
void gemm_wmma_kernel(const bf16_t* __restrict__ A, int lda,
                      const bf16_t* __restrict__ W,
                      const float* __restrict__ bias,
                      float* __restrict__ Cf, bf16_t* __restrict__ Cb, int ldc,
                      int Mrows, int K, int relu)
{
  const int lane = threadIdx.x & 31;
  const int wave = threadIdx.x >> 5;
  const int half = lane >> 4;
  const int c16  = lane & 15;
  const int nbase = blockIdx.x * NREP * 16;
  const int mbase = (blockIdx.y * 8 + wave) * 64;
  if (mbase >= Mrows) return;

  int offA[4];
#pragma unroll
  for (int r = 0; r < 4; ++r)
    offA[r] = (mbase + r * 16 + c16) * lda + half * 16;
  int offW[NREP];
#pragma unroll
  for (int n = 0; n < NREP; ++n)
    offW[n] = (nbase + n * 16 + c16) * K + half * 16;

  f32x8 acc[NREP][4];
#pragma unroll
  for (int n = 0; n < NREP; ++n)
#pragma unroll
    for (int r = 0; r < 4; ++r) acc[n][r] = zero8();

  bf16x16 bcur[NREP], acur[4];
#pragma unroll
  for (int n = 0; n < NREP; ++n) bcur[n] = load_b_frag(W, offW[n]);
#pragma unroll
  for (int r = 0; r < 4; ++r) acur[r] = load_a_frag(A, offA[r]);

  for (int k = 0; k < K; k += 32) {
    bf16x16 bnxt[NREP], anxt[4];
    const int kn = k + 32;
    if (kn < K) {
#pragma unroll
      for (int n = 0; n < NREP; ++n) bnxt[n] = load_b_frag(W, offW[n] + kn);
#pragma unroll
      for (int r = 0; r < 4; ++r) anxt[r] = load_a_frag(A, offA[r] + kn);
    }
#pragma unroll
    for (int n = 0; n < NREP; ++n)
#pragma unroll
      for (int r = 0; r < 4; ++r)
        acc[n][r] = wmma_bf16(acur[r], bcur[n], acc[n][r]);
    if (kn < K) {
#pragma unroll
      for (int n = 0; n < NREP; ++n) bcur[n] = bnxt[n];
#pragma unroll
      for (int r = 0; r < 4; ++r) acur[r] = anxt[r];
    }
  }

#pragma unroll
  for (int n = 0; n < NREP; ++n) {
    const int col = nbase + n * 16 + c16;
    const float bv = bias[col];
#pragma unroll
    for (int rep = 0; rep < 4; ++rep) {
#pragma unroll
      for (int r = 0; r < 8; ++r) {
        float v = acc[n][rep][r] + bv;
        if (relu) v = fmaxf(v, 0.0f);
        const int idx = (mbase + rep * 16 + half * 8 + r) * ldc + col;
        if (Cf) Cf[idx] = v;
        if (Cb) Cb[idx] = (bf16_t)v;
      }
    }
  }
}

// ---------------------------------------------------------------------------
// Flash attention, HD=64, one wave per (b, h, 16 query rows), bf16 in/out.
// Uniform base pointers per block; per-lane 32-bit offsets.
// ---------------------------------------------------------------------------
__global__ __launch_bounds__(32)
void attn_flash_kernel(const bf16_t* __restrict__ Qb, int ldq, long long qbs,
                       const bf16_t* __restrict__ Kb, int ldk, long long kbs,
                       const bf16_t* __restrict__ Vb, int ldv, long long vbs,
                       bf16_t* __restrict__ Ob, int ldo, long long obs,
                       int Sk)
{
  const int lane = threadIdx.x;
  const int half = lane >> 4;
  const int c16  = lane & 15;
  const int qt = blockIdx.x;
  const int h  = blockIdx.y;
  const int b  = blockIdx.z;

  const bf16_t* q  = Qb + (size_t)b * qbs + (size_t)qt * 16 * ldq + h * TTS_HD;
  const bf16_t* kp = Kb + (size_t)b * kbs + h * TTS_HD;
  const bf16_t* vp = Vb + (size_t)b * vbs + h * TTS_HD;
  bf16_t*       op = Ob + (size_t)b * obs + (size_t)qt * 16 * ldo + h * TTS_HD;

  const int qoff = c16 * ldq + half * 16;
  const bf16x16 qa0 = load_a_frag(q, qoff);
  const bf16x16 qa1 = load_a_frag(q, qoff + 32);

  f32x8 oacc[4];
#pragma unroll
  for (int dt = 0; dt < 4; ++dt) oacc[dt] = zero8();
  float mrun[8], lrun[8];
#pragma unroll
  for (int r = 0; r < 8; ++r) { mrun[r] = -1e30f; lrun[r] = 0.0f; }

  __shared__ bf16_t ptile[16][40];
  const float scale = 0.125f;  // 1/sqrt(64)

  for (int j = 0; j < Sk; j += 32) {
    // scores: two 16x16 tiles (keys j..j+15, j+16..j+31)
    f32x8 s0, s1;
    {
      const int ko = (j + c16) * ldk + half * 16;
      f32x8 c = zero8();
      c = wmma_bf16(qa0, load_b_frag(kp, ko), c);
      c = wmma_bf16(qa1, load_b_frag(kp, ko + 32), c);
      s0 = c;
    }
    {
      const int ko = (j + 16 + c16) * ldk + half * 16;
      f32x8 c = zero8();
      c = wmma_bf16(qa0, load_b_frag(kp, ko), c);
      c = wmma_bf16(qa1, load_b_frag(kp, ko + 32), c);
      s1 = c;
    }
    // online softmax (row reductions stay inside 16-lane halves: wave32)
#pragma unroll
    for (int r = 0; r < 8; ++r) {
      float v0 = s0[r] * scale, v1 = s1[r] * scale;
      float mx = fmaxf(v0, v1);
#pragma unroll
      for (int msk = 1; msk < 16; msk <<= 1) mx = fmaxf(mx, __shfl_xor(mx, msk, 32));
      const float mnew  = fmaxf(mrun[r], mx);
      const float alpha = __expf(mrun[r] - mnew);
      const float p0 = __expf(v0 - mnew);
      const float p1 = __expf(v1 - mnew);
      float rs = p0 + p1;
#pragma unroll
      for (int msk = 1; msk < 16; msk <<= 1) rs += __shfl_xor(rs, msk, 32);
      lrun[r] = lrun[r] * alpha + rs;
      mrun[r] = mnew;
      oacc[0][r] *= alpha; oacc[1][r] *= alpha;
      oacc[2][r] *= alpha; oacc[3][r] *= alpha;
      const int row = half * 8 + r;
      ptile[row][c16]      = (bf16_t)p0;
      ptile[row][c16 + 16] = (bf16_t)p1;
    }
    __syncthreads();
    // re-stripe P (C layout) -> A-operand layout via LDS
    bf16x16 pf;
#pragma unroll
    for (int e = 0; e < 16; ++e) {
      const int kk = e + (half << 4);
      pf[e] = ptile[c16][kk];
    }
    // O += P(16x32) x V(32x64). For fixed e, 16 lanes of a half read 16
    // consecutive bf16 (32B) -> hardware-coalesced.
#pragma unroll
    for (int dt = 0; dt < 4; ++dt) {
      bf16x16 vf;
      const int dcol = dt * 16 + c16;
#pragma unroll
      for (int e = 0; e < 16; ++e) {
        const int key = j + e + (half << 4);
        vf[e] = vp[key * ldv + dcol];
      }
      oacc[dt] = wmma_bf16(pf, vf, oacc[dt]);
    }
    __syncthreads();
  }

#pragma unroll
  for (int r = 0; r < 8; ++r) {
    const float inv = 1.0f / lrun[r];
    const int rbase = (half * 8 + r) * ldo;
#pragma unroll
    for (int dt = 0; dt < 4; ++dt)
      op[rbase + dt * 16 + c16] = (bf16_t)(oacc[dt][r] * inv);
  }
}

// ---------------------------------------------------------------------------
// X = LayerNorm(X + H) * g + b  (E=512, in-place f32) + bf16 copy Xb.
// ---------------------------------------------------------------------------
__global__ __launch_bounds__(256)
void add_ln512_kernel(float* __restrict__ X, bf16_t* __restrict__ Xb,
                      const float* __restrict__ Hh,
                      const float* __restrict__ g, const float* __restrict__ be)
{
  const int row = blockIdx.x;
  const int tid = threadIdx.x;
  float* x = X + (size_t)row * TTS_E;
  bf16_t* xb = Xb + (size_t)row * TTS_E;
  const float* hh = Hh + (size_t)row * TTS_E;

  const float v0 = x[tid]       + hh[tid];
  const float v1 = x[tid + 256] + hh[tid + 256];

  __shared__ float red[8];
  const int lane = tid & 31, wid = tid >> 5;

  float s = v0 + v1;
#pragma unroll
  for (int m = 1; m < 32; m <<= 1) s += __shfl_xor(s, m, 32);
  if (lane == 0) red[wid] = s;
  __syncthreads();
  const float mean =
      (red[0] + red[1] + red[2] + red[3] + red[4] + red[5] + red[6] + red[7]) *
      (1.0f / (float)TTS_E);
  __syncthreads();

  const float d0 = v0 - mean, d1 = v1 - mean;
  float qs = d0 * d0 + d1 * d1;
#pragma unroll
  for (int m = 1; m < 32; m <<= 1) qs += __shfl_xor(qs, m, 32);
  if (lane == 0) red[wid] = qs;
  __syncthreads();
  const float var =
      (red[0] + red[1] + red[2] + red[3] + red[4] + red[5] + red[6] + red[7]) *
      (1.0f / (float)TTS_E);
  const float rstd = rsqrtf(var + 1e-5f);

  const float o0 = d0 * rstd * g[tid]       + be[tid];
  const float o1 = d1 * rstd * g[tid + 256] + be[tid + 256];
  x[tid]        = o0;
  x[tid + 256]  = o1;
  xb[tid]       = (bf16_t)o0;
  xb[tid + 256] = (bf16_t)o1;
}

// ---------------------------------------------------------------------------
__global__ __launch_bounds__(256)
void embed_kernel(const int* __restrict__ text, const float* __restrict__ emb,
                  float* __restrict__ X, bf16_t* __restrict__ Xb)
{
  const int bs = blockIdx.x;
  const int tok = text[bs];
  const float* src = emb + (size_t)tok * TTS_E;
  float* dst = X + (size_t)bs * TTS_E;
  bf16_t* dstb = Xb + (size_t)bs * TTS_E;
  const float a = src[threadIdx.x];
  const float b = src[threadIdx.x + 256];
  dst[threadIdx.x]        = a;
  dst[threadIdx.x + 256]  = b;
  dstb[threadIdx.x]       = (bf16_t)a;
  dstb[threadIdx.x + 256] = (bf16_t)b;
}

__global__ void cumsum_kernel(const int* __restrict__ dur, int* __restrict__ cum)
{
  const int b = blockIdx.x;
  if (threadIdx.x == 0) {
    int s = 0;
    for (int j = 0; j < TTS_S; ++j) { s += dur[b * TTS_S + j]; cum[b * TTS_S + j] = s; }
  }
}

// Length regulator: Y[b,t,:] = enc[b, searchsorted_right(cum[b], t), :]
__global__ __launch_bounds__(256)
void regulate_kernel(const float* __restrict__ enc, const int* __restrict__ cum,
                     float* __restrict__ Y, bf16_t* __restrict__ Yb16)
{
  const int t = blockIdx.x;
  const int b = blockIdx.y;
  const int* c = cum + b * TTS_S;
  int lo = 0, hi = TTS_S;
  while (lo < hi) {
    const int mid = (lo + hi) >> 1;
    if (c[mid] <= t) lo = mid + 1; else hi = mid;
  }
  const int idx = lo < (TTS_S - 1) ? lo : (TTS_S - 1);
  const float* src = enc + ((size_t)b * TTS_S + idx) * TTS_E;
  float* dst = Y + ((size_t)b * TTS_T + t) * TTS_E;
  bf16_t* dstb = Yb16 + ((size_t)b * TTS_T + t) * TTS_E;
  const float a = src[threadIdx.x];
  const float d = src[threadIdx.x + 256];
  dst[threadIdx.x]        = a;
  dst[threadIdx.x + 256]  = d;
  dstb[threadIdx.x]       = (bf16_t)a;
  dstb[threadIdx.x + 256] = (bf16_t)d;
}

// dur_pred[row] = dot(Hd[row,:HID], w2) + b2
__global__ __launch_bounds__(256)
void durpred_kernel(const float* __restrict__ Hd, const float* __restrict__ w2,
                    const float* __restrict__ b2, float* __restrict__ out)
{
  const int row = blockIdx.x;
  const float* h = Hd + (size_t)row * TTS_HID;
  float s = 0.0f;
  for (int j = threadIdx.x; j < TTS_HID; j += 256) s += h[j] * w2[j];
#pragma unroll
  for (int m = 1; m < 32; m <<= 1) s += __shfl_xor(s, m, 32);
  __shared__ float red[8];
  const int lane = threadIdx.x & 31, wid = threadIdx.x >> 5;
  if (lane == 0) red[wid] = s;
  __syncthreads();
  if (threadIdx.x == 0) {
    out[row] = red[0] + red[1] + red[2] + red[3] +
               red[4] + red[5] + red[6] + red[7] + b2[0];
  }
}

// f32 -> bf16, 4 elems/thread, n must be a multiple of 1024.
__global__ __launch_bounds__(256)
void cvt_bf16_kernel(const float* __restrict__ src, bf16_t* __restrict__ dst)
{
  const size_t i = ((size_t)blockIdx.x * 256 + threadIdx.x) * 4;
  const f32x4 v = *reinterpret_cast<const f32x4*>(src + i);
  bf16x4 o;
  o[0] = (bf16_t)v.x; o[1] = (bf16_t)v.y; o[2] = (bf16_t)v.z; o[3] = (bf16_t)v.w;
  *reinterpret_cast<bf16x4*>(dst + i) = o;
}

// ---------------------------------------------------------------------------
extern "C" void kernel_launch(void* const* d_in, const int* in_sizes, int n_in,
                              void* d_out, int out_size, void* d_ws, size_t ws_size,
                              hipStream_t stream)
{
  (void)in_sizes; (void)n_in; (void)out_size; (void)ws_size;

  const int*   text      = (const int*)d_in[0];
  const float* spec      = (const float*)d_in[1];
  const int*   durations = (const int*)d_in[2];
  const float* emb       = (const float*)d_in[3];
  const float* enc_Wqkv  = (const float*)d_in[4];
  const float* enc_bqkv  = (const float*)d_in[5];
  const float* enc_Wo    = (const float*)d_in[6];
  const float* enc_bo    = (const float*)d_in[7];
  const float* enc_W1    = (const float*)d_in[8];
  const float* enc_b1    = (const float*)d_in[9];
  const float* enc_W2    = (const float*)d_in[10];
  const float* enc_b2    = (const float*)d_in[11];
  const float* enc_ln1g  = (const float*)d_in[12];
  const float* enc_ln1b  = (const float*)d_in[13];
  const float* enc_ln2g  = (const float*)d_in[14];
  const float* enc_ln2b  = (const float*)d_in[15];
  const float* sa_Wqkv   = (const float*)d_in[16];
  const float* sa_bqkv   = (const float*)d_in[17];
  const float* sa_Wo     = (const float*)d_in[18];
  const float* sa_bo     = (const float*)d_in[19];
  const float* ca_Wqkv   = (const float*)d_in[20];
  const float* ca_bqkv   = (const float*)d_in[21];
  const float* ca_Wo     = (const float*)d_in[22];
  const float* ca_bo     = (const float*)d_in[23];
  const float* dec_W1    = (const float*)d_in[24];
  const float* dec_b1    = (const float*)d_in[25];
  const float* dec_W2    = (const float*)d_in[26];
  const float* dec_b2    = (const float*)d_in[27];
  const float* dec_ln1g  = (const float*)d_in[28];
  const float* dec_ln1b  = (const float*)d_in[29];
  const float* dec_ln2g  = (const float*)d_in[30];
  const float* dec_ln2b  = (const float*)d_in[31];
  const float* dec_ln3g  = (const float*)d_in[32];
  const float* dec_ln3b  = (const float*)d_in[33];
  const float* dp_W1     = (const float*)d_in[34];
  const float* dp_b1     = (const float*)d_in[35];
  const float* dp_W2     = (const float*)d_in[36];
  const float* dp_b2     = (const float*)d_in[37];
  const float* fc_W      = (const float*)d_in[38];
  const float* fc_b      = (const float*)d_in[39];

  float* outp = (float*)d_out;

  const int E  = TTS_E, E3 = 3 * TTS_E;
  const int BS = TTS_B * TTS_S;   // 2048
  const int BT = TTS_B * TTS_T;   // 8192
  const int BM = TTS_B * TTS_M;   // 4096

  // ---- workspace layout (units: float slots) ----
  float* wsf = (float*)d_ws;
  size_t off = 0;
  float*  Ax   = wsf + off;                 off += (size_t)BS * E;           // enc act f32
  float*  Yb   = wsf + off;                 off += (size_t)BT * E;           // dec act f32 (also dp hidden)
  float*  Hf   = wsf + off;                 off += (size_t)BT * E;           // proj/ffn2 f32 temp
  bf16_t* Axb  = (bf16_t*)(wsf + off);      off += (size_t)BS * E / 2;       // enc act bf16
  bf16_t* Ybb  = (bf16_t*)(wsf + off);      off += (size_t)BT * E / 2;       // dec act bf16
  bf16_t* Hbb  = (bf16_t*)(wsf + off);      off += (size_t)BT * E / 2;       // attn out bf16
  bf16_t* Gbf  = (bf16_t*)(wsf + off);      off += (size_t)BT * TTS_HID / 2; // qkv / ffn-hidden / cross-q bf16
  bf16_t* KVb  = (bf16_t*)(wsf + off);      off += (size_t)BM * 2 * E / 2;   // cross K,V bf16
  bf16_t* spcb = (bf16_t*)(wsf + off);      off += (size_t)BM * E / 2;       // spectrogram bf16
  int*    cum  = (int*)(wsf + off);         off += (size_t)BS;

  // bf16 weight staging
  bf16_t* wb = (bf16_t*)(wsf + off);
  bf16_t* enc_Wqkv_b = wb;                               wb += (size_t)TTS_L * E3 * E;
  bf16_t* enc_Wo_b   = wb;                               wb += (size_t)TTS_L * E * E;
  bf16_t* enc_W1_b   = wb;                               wb += (size_t)TTS_L * TTS_HID * E;
  bf16_t* enc_W2_b   = wb;                               wb += (size_t)TTS_L * E * TTS_HID;
  bf16_t* sa_Wqkv_b  = wb;                               wb += (size_t)TTS_L * E3 * E;
  bf16_t* sa_Wo_b    = wb;                               wb += (size_t)TTS_L * E * E;
  bf16_t* ca_Wqkv_b  = wb;                               wb += (size_t)TTS_L * E3 * E;
  bf16_t* ca_Wo_b    = wb;                               wb += (size_t)TTS_L * E * E;
  bf16_t* dec_W1_b   = wb;                               wb += (size_t)TTS_L * TTS_HID * E;
  bf16_t* dec_W2_b   = wb;                               wb += (size_t)TTS_L * E * TTS_HID;
  bf16_t* dp_W1_b    = wb;                               wb += (size_t)TTS_HID * E;
  bf16_t* fc_W_b     = wb;                               wb += (size_t)TTS_OUT * E;

  // ---- stage weights + spectrogram in bf16 (n % 1024 == 0 for all) ----
  auto CVT = [&](const float* s, bf16_t* d, size_t n) {
    cvt_bf16_kernel<<<(unsigned)(n / 1024), 256, 0, stream>>>(s, d);
  };
  CVT(enc_Wqkv, enc_Wqkv_b, (size_t)TTS_L * E3 * E);
  CVT(enc_Wo,   enc_Wo_b,   (size_t)TTS_L * E * E);
  CVT(enc_W1,   enc_W1_b,   (size_t)TTS_L * TTS_HID * E);
  CVT(enc_W2,   enc_W2_b,   (size_t)TTS_L * E * TTS_HID);
  CVT(sa_Wqkv,  sa_Wqkv_b,  (size_t)TTS_L * E3 * E);
  CVT(sa_Wo,    sa_Wo_b,    (size_t)TTS_L * E * E);
  CVT(ca_Wqkv,  ca_Wqkv_b,  (size_t)TTS_L * E3 * E);
  CVT(ca_Wo,    ca_Wo_b,    (size_t)TTS_L * E * E);
  CVT(dec_W1,   dec_W1_b,   (size_t)TTS_L * TTS_HID * E);
  CVT(dec_W2,   dec_W2_b,   (size_t)TTS_L * E * TTS_HID);
  CVT(dp_W1,    dp_W1_b,    (size_t)TTS_HID * E);
  CVT(fc_W,     fc_W_b,     (size_t)TTS_OUT * E);
  CVT(spec,     spcb,       (size_t)BM * E);

  const long long sE3 = (long long)TTS_S * E3;
  const long long sE  = (long long)TTS_S * E;
  const long long tE3 = (long long)TTS_T * E3;
  const long long tE  = (long long)TTS_T * E;
  const long long m2E = (long long)TTS_M * 2 * E;

  // ---- embedding ----
  embed_kernel<<<BS, 256, 0, stream>>>(text, emb, Ax, Axb);

  // ---- encoder (rows = BS = 2048, M grid = BS/512 = 4) ----
  for (int i = 0; i < TTS_L; ++i) {
    gemm_wmma_kernel<2><<<dim3(E3 / 32, BS / 512), 256, 0, stream>>>(
        Axb, E, enc_Wqkv_b + (size_t)i * E3 * E, enc_bqkv + (size_t)i * E3,
        nullptr, Gbf, E3, BS, E, 0);
    attn_flash_kernel<<<dim3(TTS_S / 16, TTS_H, TTS_B), 32, 0, stream>>>(
        Gbf, E3, sE3, Gbf + E, E3, sE3, Gbf + 2 * E, E3, sE3,
        Hbb, E, sE, TTS_S);
    gemm_wmma_kernel<2><<<dim3(E / 32, BS / 512), 256, 0, stream>>>(
        Hbb, E, enc_Wo_b + (size_t)i * E * E, enc_bo + (size_t)i * E,
        Hf, nullptr, E, BS, E, 0);
    add_ln512_kernel<<<BS, 256, 0, stream>>>(
        Ax, Axb, Hf, enc_ln1g + (size_t)i * E, enc_ln1b + (size_t)i * E);
    gemm_wmma_kernel<2><<<dim3(TTS_HID / 32, BS / 512), 256, 0, stream>>>(
        Axb, E, enc_W1_b + (size_t)i * TTS_HID * E, enc_b1 + (size_t)i * TTS_HID,
        nullptr, Gbf, TTS_HID, BS, E, 1);
    gemm_wmma_kernel<2><<<dim3(E / 32, BS / 512), 256, 0, stream>>>(
        Gbf, TTS_HID, enc_W2_b + (size_t)i * E * TTS_HID, enc_b2 + (size_t)i * E,
        Hf, nullptr, E, BS, TTS_HID, 0);
    add_ln512_kernel<<<BS, 256, 0, stream>>>(
        Ax, Axb, Hf, enc_ln2g + (size_t)i * E, enc_ln2b + (size_t)i * E);
  }

  // ---- duration predictor (f32 hidden staged in Yb, free at this point) ----
  gemm_wmma_kernel<2><<<dim3(TTS_HID / 32, BS / 512), 256, 0, stream>>>(
      Axb, E, dp_W1_b, dp_b1, Yb, nullptr, TTS_HID, BS, E, 1);
  durpred_kernel<<<BS, 256, 0, stream>>>(
      Yb, dp_W2, dp_b2, outp + (size_t)BT * TTS_OUT);

  // ---- length regulator ----
  cumsum_kernel<<<TTS_B, 32, 0, stream>>>(durations, cum);
  regulate_kernel<<<dim3(TTS_T, TTS_B), 256, 0, stream>>>(Ax, cum, Yb, Ybb);

  // ---- decoder (rows = BT = 8192, M grid = 16) ----
  for (int i = 0; i < TTS_L; ++i) {
    // self-attention
    gemm_wmma_kernel<2><<<dim3(E3 / 32, BT / 512), 256, 0, stream>>>(
        Ybb, E, sa_Wqkv_b + (size_t)i * E3 * E, sa_bqkv + (size_t)i * E3,
        nullptr, Gbf, E3, BT, E, 0);
    attn_flash_kernel<<<dim3(TTS_T / 16, TTS_H, TTS_B), 32, 0, stream>>>(
        Gbf, E3, tE3, Gbf + E, E3, tE3, Gbf + 2 * E, E3, tE3,
        Hbb, E, tE, TTS_T);
    gemm_wmma_kernel<2><<<dim3(E / 32, BT / 512), 256, 0, stream>>>(
        Hbb, E, sa_Wo_b + (size_t)i * E * E, sa_bo + (size_t)i * E,
        Hf, nullptr, E, BT, E, 0);
    add_ln512_kernel<<<BT, 256, 0, stream>>>(
        Yb, Ybb, Hf, dec_ln1g + (size_t)i * E, dec_ln1b + (size_t)i * E);

    // cross-attention vs spectrogram memory
    gemm_wmma_kernel<2><<<dim3(E / 32, BT / 512), 256, 0, stream>>>(
        Ybb, E, ca_Wqkv_b + (size_t)i * E3 * E, ca_bqkv + (size_t)i * E3,
        nullptr, Gbf, E, BT, E, 0);                                    // Q
    gemm_wmma_kernel<2><<<dim3(2 * E / 32, BM / 512), 256, 0, stream>>>(
        spcb, E, ca_Wqkv_b + (size_t)i * E3 * E + (size_t)E * E,
        ca_bqkv + (size_t)i * E3 + E, nullptr, KVb, 2 * E, BM, E, 0);  // K,V
    attn_flash_kernel<<<dim3(TTS_T / 16, TTS_H, TTS_B), 32, 0, stream>>>(
        Gbf, E, tE, KVb, 2 * E, m2E, KVb + E, 2 * E, m2E,
        Hbb, E, tE, TTS_M);
    gemm_wmma_kernel<2><<<dim3(E / 32, BT / 512), 256, 0, stream>>>(
        Hbb, E, ca_Wo_b + (size_t)i * E * E, ca_bo + (size_t)i * E,
        Hf, nullptr, E, BT, E, 0);
    add_ln512_kernel<<<BT, 256, 0, stream>>>(
        Yb, Ybb, Hf, dec_ln2g + (size_t)i * E, dec_ln2b + (size_t)i * E);

    // FFN
    gemm_wmma_kernel<2><<<dim3(TTS_HID / 32, BT / 512), 256, 0, stream>>>(
        Ybb, E, dec_W1_b + (size_t)i * TTS_HID * E, dec_b1 + (size_t)i * TTS_HID,
        nullptr, Gbf, TTS_HID, BT, E, 1);
    gemm_wmma_kernel<2><<<dim3(E / 32, BT / 512), 256, 0, stream>>>(
        Gbf, TTS_HID, dec_W2_b + (size_t)i * E * TTS_HID, dec_b2 + (size_t)i * E,
        Hf, nullptr, E, BT, TTS_HID, 0);
    add_ln512_kernel<<<BT, 256, 0, stream>>>(
        Yb, Ybb, Hf, dec_ln3g + (size_t)i * E, dec_ln3b + (size_t)i * E);
  }

  // ---- final projection (first output, N=80 -> NREP=1) ----
  gemm_wmma_kernel<1><<<dim3(TTS_OUT / 16, BT / 512), 256, 0, stream>>>(
      Ybb, E, fc_W_b, fc_b, outp, nullptr, TTS_OUT, BT, E, 0);
}
